// AttentionBlock_58454504899066
// MI455X (gfx1250) — matrix operand order repaired and checked by reading, not compiled, not run on hardware
//
#include <hip/hip_runtime.h>
#include <hip/hip_bf16.h>

#define DEVINL __device__ __forceinline__

typedef __attribute__((ext_vector_type(16))) __bf16 v16bf;
typedef __attribute__((ext_vector_type(8)))  __bf16 v8bf;
typedef __attribute__((ext_vector_type(8)))  float  v8f;

#define BATCH 2
#define CCH   512
#define TT    4096
#define NH    8
#define CH    64
#define OC3   1536
#define EPSV  1e-8f

// ---------------- fragment helpers (layouts per CDNA5 ISA §7.12.2) --

DEVINL v16bf frag_from(const __bf16* p0, const __bf16* p1) {
  v8bf lo = *reinterpret_cast<const v8bf*>(p0);
  v8bf hi = *reinterpret_cast<const v8bf*>(p1);
  v16bf r;
#pragma unroll
  for (int i = 0; i < 8; ++i) { r[i] = lo[i]; r[i + 8] = hi[i]; }
  return r;
}

// A fragment: 16x32 (MxK), source row-major, leading dim ld.
DEVINL v16bf fragA(const __bf16* src, int ld, int r0, int k0, int lane) {
  int hl = lane >> 4, lr = lane & 15;
  const __bf16* p = src + (size_t)(r0 + lr) * ld + k0 + hl * 8;
  return frag_from(p, p + 16);
}

// B fragment: 32x16 (KxN), element (k,n) at src[n*ld + k].
DEVINL v16bf fragB(const __bf16* src, int ld, int n0, int k0, int lane) {
  int hl = lane >> 4, lr = lane & 15;
  const __bf16* p = src + (size_t)(n0 + lr) * ld + k0 + hl * 8;
  return frag_from(p, p + 16);
}

DEVINL v8f wmma_bf16(v16bf a, v16bf b, v8f c) {
  return __builtin_amdgcn_wmma_f32_16x16x32_bf16(false, a, false, b, (short)0, c,
                                                 false, false);
}

DEVINL v8f vzero8() {
  v8f z = {0.f, 0.f, 0.f, 0.f, 0.f, 0.f, 0.f, 0.f};
  return z;
}

// Async global->LDS 16-byte copy (GV mode, ASYNCcnt-tracked). Generic LDS
// pointer's low 32 bits are the wave-relative LDS address.
DEVINL void async_b128(const __bf16* g, __bf16* l) {
  unsigned loff = (unsigned)(size_t)l;
  asm volatile("global_load_async_to_lds_b128 %0, %1, off"
               :: "v"(loff), "v"(g) : "memory");
}

// ---------------- kernel 0: fp32 weights -> bf16 ------------------------------

__global__ void cvt_weights_kernel(const float* __restrict__ wq, const float* __restrict__ wp,
                                   __bf16* __restrict__ oq, __bf16* __restrict__ op,
                                   int nq, int np) {
  int i = blockIdx.x * blockDim.x + threadIdx.x;
  if (i < nq) oq[i] = (__bf16)wq[i];
  if (i < np) op[i] = (__bf16)wp[i];
}

// ---------------- kernel 1: RMS norm, write xn as (b, t, c) bf16 --------------

__global__ void rmsnorm_kernel(const float* __restrict__ x, const float* __restrict__ nw,
                               __bf16* __restrict__ xn) {
  int tid = blockIdx.x * blockDim.x + threadIdx.x;       // 0 .. B*T-1
  int b = tid / TT, t = tid % TT;
  const float* xp = x + (size_t)b * CCH * TT + t;
  float ss = 0.f;
#pragma unroll 8
  for (int c = 0; c < CCH; ++c) { float v = xp[(size_t)c * TT]; ss += v * v; }
  float inv = rsqrtf(ss / (float)CCH + EPSV) * 22.627416997969522f; // * sqrt(512)
  __bf16* op = xn + (size_t)tid * CCH;
#pragma unroll 8
  for (int c = 0; c < CCH; ++c)
    op[c] = (__bf16)(xp[(size_t)c * TT] * inv * nw[c]);
}

// ---------------- kernel 2: QKV GEMM, 64x32 per wave (4x2 WMMA tiles) ---------

__global__ void __launch_bounds__(256, 1)
qkv_gemm_kernel(const __bf16* __restrict__ w, const float* __restrict__ bias,
                const __bf16* __restrict__ xn,
                __bf16* __restrict__ qT, __bf16* __restrict__ kT,
                __bf16* __restrict__ vT) {
  int wid  = (blockIdx.x * blockDim.x + threadIdx.x) >> 5;
  int lane = threadIdx.x & 31;
  int b  = wid / (24 * 128);
  int r  = wid % (24 * 128);
  int o0 = (r >> 7) * 64;
  int t0 = (r & 127) * 32;
  const __bf16* xb = xn + (size_t)b * TT * CCH;

  v8f acc[4][2];
#pragma unroll
  for (int i = 0; i < 4; ++i)
#pragma unroll
    for (int j = 0; j < 2; ++j) acc[i][j] = vzero8();

  for (int k0 = 0; k0 < CCH; k0 += 32) {
    if (k0 + 32 < CCH) {
      __builtin_prefetch(w  + (size_t)o0 * CCH + k0 + 32, 0, 3);
      __builtin_prefetch(xb + (size_t)t0 * CCH + k0 + 32, 0, 3);
    }
    v16bf af[4], bf[2];
#pragma unroll
    for (int i = 0; i < 4; ++i) af[i] = fragA(w,  CCH, o0 + i * 16, k0, lane);
#pragma unroll
    for (int j = 0; j < 2; ++j) bf[j] = fragB(xb, CCH, t0 + j * 16, k0, lane);
#pragma unroll
    for (int i = 0; i < 4; ++i)
#pragma unroll
      for (int j = 0; j < 2; ++j) acc[i][j] = wmma_bf16(af[i], bf[j], acc[i][j]);
  }

  int hl = lane >> 4, lr = lane & 15;
  int sect = o0 >> 9;                 // 0=q, 1=k, 2=v (64-row tiles are head-aligned)
  int bh   = b * NH + ((o0 >> 6) & 7);
#pragma unroll
  for (int i = 0; i < 4; ++i)
#pragma unroll
    for (int j = 0; j < 2; ++j)
#pragma unroll
      for (int v = 0; v < 8; ++v) {
        int cc = i * 16 + v + 8 * hl;          // channel within head
        int t  = t0 + j * 16 + lr;
        float val = acc[i][j][v] + bias[o0 + cc];
        if (sect == 0)          // q : (bh, t, ch), fold both ch^-0.25 scales
          qT[((size_t)bh * TT + t) * CH + cc] = (__bf16)(val * 0.125f);
        else if (sect == 1)     // k : (bh, t, ch)
          kT[((size_t)bh * TT + t) * CH + cc] = (__bf16)val;
        else                    // v : (bh, ch, t)
          vT[((size_t)bh * CH + cc) * TT + t] = (__bf16)val;
      }
}

// ---------------- kernel 3: flash attention -----------------------------------
// Block = 8 waves sharing one (batch,head); each wave owns 16 query rows.
// K/V 32-wide tiles staged cooperatively in LDS via async-to-LDS, double-buffered.

__global__ void __launch_bounds__(256, 1)
attn_kernel(const __bf16* __restrict__ qT, const __bf16* __restrict__ kT,
            const __bf16* __restrict__ vT, __bf16* __restrict__ ab) {
  __shared__ __bf16 kbuf[2][32 * 64];   // (s_local, ch)   4 KB each
  __shared__ __bf16 vbuf[2][64 * 32];   // (ch, s_local)   4 KB each
  __shared__ __bf16 pbuf[8][16 * 32];   // per-wave P tile 1 KB each

  int tid   = threadIdx.x;
  int wslot = tid >> 5, lane = tid & 31;
  int hl = lane >> 4, lr = lane & 15;
  int bh = blockIdx.x >> 5;
  int q0 = ((blockIdx.x & 31) * 8 + wslot) * 16;

  const __bf16* qb = qT + (size_t)bh * TT * CH;
  const __bf16* kb = kT + (size_t)bh * TT * CH;
  const __bf16* vb = vT + (size_t)bh * CH * TT;

  int vc = tid >> 2, vs = tid & 3;      // V-tile copy mapping: row c, 16B segment

  auto fill = [&](int buf, int s0) {
    // K tile: rows s0..s0+31 are contiguous 4 KB in (t, ch) layout
    async_b128(kb + (size_t)s0 * CH + tid * 8, &kbuf[buf][tid * 8]);
    // V tile: 64 rows of 64 B, stride T elements in (ch, t) layout
    async_b128(vb + (size_t)vc * TT + s0 + vs * 8, &vbuf[buf][tid * 8]);
  };

  v16bf qf0 = fragA(qb, CH, q0, 0,  lane);
  v16bf qf1 = fragA(qb, CH, q0, 32, lane);

  v8f o0 = vzero8(), o1 = vzero8(), o2 = vzero8(), o3 = vzero8();
  float m[8], l[8];
#pragma unroll
  for (int v = 0; v < 8; ++v) { m[v] = -1e30f; l[v] = 0.f; }

  __bf16* pw = pbuf[wslot];

  fill(0, 0);
  asm volatile("s_wait_asynccnt 0x0" ::: "memory");
  __syncthreads();

  for (int it = 0; it < TT / 32; ++it) {
    int s0  = it * 32;
    int cur = it & 1;
    if (it + 1 < TT / 32) fill(cur ^ 1, s0 + 32);

    const __bf16* kt = kbuf[cur];
    const __bf16* vt = vbuf[cur];

    // S = Q K^T (two 16-wide key sub-tiles; K dim = 64 -> 2 wmmas each)
    v8f S0 = vzero8(), S1 = vzero8();
    S0 = wmma_bf16(qf0, fragB(kt, CH, 0,  0,  lane), S0);
    S0 = wmma_bf16(qf1, fragB(kt, CH, 0,  32, lane), S0);
    S1 = wmma_bf16(qf0, fragB(kt, CH, 16, 0,  lane), S1);
    S1 = wmma_bf16(qf1, fragB(kt, CH, 16, 32, lane), S1);

    // online softmax; row m lives in VGPR index across each 16-lane half
#pragma unroll
    for (int v = 0; v < 8; ++v) {
      float rm = fmaxf(S0[v], S1[v]);
      rm = fmaxf(rm, __shfl_xor(rm, 1, 32));
      rm = fmaxf(rm, __shfl_xor(rm, 2, 32));
      rm = fmaxf(rm, __shfl_xor(rm, 4, 32));
      rm = fmaxf(rm, __shfl_xor(rm, 8, 32));
      float mn = fmaxf(m[v], rm);
      float al = __expf(m[v] - mn);
      m[v] = mn;
      float p0 = __expf(S0[v] - mn);
      float p1 = __expf(S1[v] - mn);
      S0[v] = p0; S1[v] = p1;
      float rs = p0 + p1;
      rs += __shfl_xor(rs, 1, 32);
      rs += __shfl_xor(rs, 2, 32);
      rs += __shfl_xor(rs, 4, 32);
      rs += __shfl_xor(rs, 8, 32);
      l[v] = l[v] * al + rs;
      o0[v] *= al; o1[v] *= al; o2[v] *= al; o3[v] *= al;
    }

    // C-layout -> A-layout transpose through wave-private LDS (in-order DS)
#pragma unroll
    for (int v = 0; v < 8; ++v) {
      int row = v + 8 * hl;
      pw[row * 32 + lr]      = (__bf16)S0[v];
      pw[row * 32 + 16 + lr] = (__bf16)S1[v];
    }
    asm volatile("s_wait_dscnt 0" ::: "memory");
    v16bf pf = fragA(pw, 32, 0, 0, lane);
    asm volatile("" ::: "memory");

    // O += P (16x32) @ V-tile (32 x 64)
    o0 = wmma_bf16(pf, fragB(vt, 32, 0,  0, lane), o0);
    o1 = wmma_bf16(pf, fragB(vt, 32, 16, 0, lane), o1);
    o2 = wmma_bf16(pf, fragB(vt, 32, 32, 0, lane), o2);
    o3 = wmma_bf16(pf, fragB(vt, 32, 48, 0, lane), o3);

    asm volatile("s_wait_asynccnt 0x0" ::: "memory");  // own async writes landed
    __syncthreads();                                   // publish next buffer
  }

  // epilogue: a stored as (b, t, c) bf16 for the proj GEMM B-operand
  int b = bh >> 3, head = bh & 7;
#pragma unroll
  for (int v = 0; v < 8; ++v) {
    int t = q0 + v + 8 * hl;
    float inv = 1.f / l[v];
    size_t base = ((size_t)b * TT + t) * CCH + head * CH + lr;
    ab[base + 0]  = (__bf16)(o0[v] * inv);
    ab[base + 16] = (__bf16)(o1[v] * inv);
    ab[base + 32] = (__bf16)(o2[v] * inv);
    ab[base + 48] = (__bf16)(o3[v] * inv);
  }
}

// ---------------- kernel 4: proj GEMM + bias + residual, fp32 out -------------

__global__ void __launch_bounds__(256, 1)
proj_gemm_kernel(const __bf16* __restrict__ w, const float* __restrict__ bias,
                 const __bf16* __restrict__ ab, const float* __restrict__ x,
                 float* __restrict__ out) {
  int wid  = (blockIdx.x * blockDim.x + threadIdx.x) >> 5;
  int lane = threadIdx.x & 31;
  int b  = wid / (8 * 128);
  int r  = wid % (8 * 128);
  int o0 = (r >> 7) * 64;
  int t0 = (r & 127) * 32;
  const __bf16* bb = ab + (size_t)b * TT * CCH;

  v8f acc[4][2];
#pragma unroll
  for (int i = 0; i < 4; ++i)
#pragma unroll
    for (int j = 0; j < 2; ++j) acc[i][j] = vzero8();

  for (int k0 = 0; k0 < CCH; k0 += 32) {
    if (k0 + 32 < CCH) {
      __builtin_prefetch(w  + (size_t)o0 * CCH + k0 + 32, 0, 3);
      __builtin_prefetch(bb + (size_t)t0 * CCH + k0 + 32, 0, 3);
    }
    v16bf af[4], bf[2];
#pragma unroll
    for (int i = 0; i < 4; ++i) af[i] = fragA(w,  CCH, o0 + i * 16, k0, lane);
#pragma unroll
    for (int j = 0; j < 2; ++j) bf[j] = fragB(bb, CCH, t0 + j * 16, k0, lane);
#pragma unroll
    for (int i = 0; i < 4; ++i)
#pragma unroll
      for (int j = 0; j < 2; ++j) acc[i][j] = wmma_bf16(af[i], bf[j], acc[i][j]);
  }

  int hl = lane >> 4, lr = lane & 15;
#pragma unroll
  for (int i = 0; i < 4; ++i)
#pragma unroll
    for (int j = 0; j < 2; ++j)
#pragma unroll
      for (int v = 0; v < 8; ++v) {
        int o = o0 + i * 16 + v + 8 * hl;
        int t = t0 + j * 16 + lr;
        size_t idx = ((size_t)b * CCH + o) * TT + t;
        out[idx] = acc[i][j][v] + bias[o] + x[idx];   // + residual h
      }
}

// ---------------- host launcher ----------------------------------------------

extern "C" void kernel_launch(void* const* d_in, const int* in_sizes, int n_in,
                              void* d_out, int out_size, void* d_ws, size_t ws_size,
                              hipStream_t stream) {
  const float* x      = (const float*)d_in[0];
  const float* nw     = (const float*)d_in[1];
  const float* qkv_w  = (const float*)d_in[2];
  const float* qkv_b  = (const float*)d_in[3];
  const float* proj_w = (const float*)d_in[4];
  const float* proj_b = (const float*)d_in[5];
  float* out = (float*)d_out;

  char* ws = (char*)d_ws;
  __bf16* qkvw_bf  = (__bf16*)(ws);                 // 1,572,864 B
  __bf16* projw_bf = (__bf16*)(ws + 1572864);       //   524,288 B
  __bf16* xn       = (__bf16*)(ws + 2097152);       // 8,388,608 B
  __bf16* qT       = (__bf16*)(ws + 10485760);      // 8,388,608 B
  __bf16* kT       = (__bf16*)(ws + 18874368);      // 8,388,608 B
  __bf16* vT       = (__bf16*)(ws + 27262976);      // 8,388,608 B
  __bf16* ab       = (__bf16*)(ws + 35651584);      // 8,388,608 B -> 44,040,192 total

  cvt_weights_kernel<<<3072, 256, 0, stream>>>(qkv_w, proj_w, qkvw_bf, projw_bf,
                                               OC3 * CCH, CCH * CCH);
  rmsnorm_kernel<<<32, 256, 0, stream>>>(x, nw, xn);
  // QKV GEMM: 2 * 24 * 128 = 6144 waves, 8 waves/block
  qkv_gemm_kernel<<<768, 256, 0, stream>>>(qkvw_bf, qkv_b, xn, qT, kT, vT);
  // attention: 16 bh * 32 q-blocks = 512 blocks of 8 cooperating waves
  attn_kernel<<<512, 256, 0, stream>>>(qT, kT, vT, ab);
  // proj GEMM: 2 * 8 * 128 = 2048 waves
  proj_gemm_kernel<<<256, 256, 0, stream>>>(projw_bf, proj_b, ab, x, out);
}
